// ConvPMF_47906065219740
// MI455X (gfx1250) — matrix operand order repaired and checked by reading, not compiled, not run on hardware
//
#include <hip/hip_runtime.h>
#include <hip/hip_bf16.h>

// ---------------- problem constants (from reference setup) ----------------
#define PB   128      // batch
#define PR   16       // reviews per doc
#define PL   256      // tokens per review
#define PF   32       // n_factor (conv filters)
#define PE   128      // embed dim
#define PW   5        // conv window
#define PBR  (PB*PR)  // 2048 reviews

// LDS row padding: 136 halfwords = 272B -> rotates across 64 banks
#define XPITCH 136
#define FMPITCH 33

typedef __attribute__((ext_vector_type(16))) __bf16 bfvec16;
typedef __attribute__((ext_vector_type(8)))  float  fvec8;

union FragU { uint4 q[2]; bfvec16 v; };

__device__ __forceinline__ unsigned short f2bf(float f) {
    unsigned u = __builtin_bit_cast(unsigned, f);
    // round-to-nearest-even truncation to bf16
    return (unsigned short)((u + 0x7FFFu + ((u >> 16) & 1u)) >> 16);
}

// ---------------------------------------------------------------------------
// Kernel 1: one workgroup (128 threads = 4 wave32) per review.
//  - gather token embeddings -> LDS bf16 with zero halo (conv 'same' padding)
//  - conv as 5 shifted GEMMs on v_wmma_f32_16x16x32_bf16, f32 accum
//  - tanh -> fm tile in LDS
//  - per (review,filter): max over L  -> maxbuf[br*32+f]
//                         entropy row -> summed over f into entbuf[br]
// ---------------------------------------------------------------------------
__global__ __launch_bounds__(128) void convpmf_main(
    const int* __restrict__ docs,        // [PBR, PL]
    const float* __restrict__ embed,     // [V, PE]
    const float* __restrict__ conv_w,    // [PF, PE, PW]
    float* __restrict__ maxbuf,          // [PBR, PF]
    float* __restrict__ entbuf)          // [PBR]
{
    extern __shared__ __align__(16) unsigned char smem[];
    int*            tok = (int*)(smem);                               // 1024 B
    unsigned short* xs  = (unsigned short*)(smem + 1024);             // 260*136*2 = 70720
    unsigned short* wb  = (unsigned short*)(smem + 1024 + 70720);     // 5*32*136*2 = 43520
    float*          fm  = (float*)(smem + 1024 + 70720 + 43520);      // 256*33*4 = 33792
    float*          pmax= (float*)(smem + 1024 + 70720 + 43520 + 33792);        // 4*32*4
    float*          ps  = pmax + 128;
    float*          pt  = ps + 128;

    const int br   = blockIdx.x;        // review index 0..2047
    const int t    = threadIdx.x;       // 0..127
    const int lane = t & 31;
    const int wave = t >> 5;
    const int half = lane >> 4;         // 0/1 : K-half selector per WMMA layout
    const int mrow = lane & 15;         // M (or N) index within tile

    // ---- load tokens + zero halo rows (rows 0,1 and 258,259) ----
    #pragma unroll
    for (int i = t; i < PL; i += 128) tok[i] = docs[(size_t)br * PL + i];
    for (int i = t; i < 4 * XPITCH; i += 128) {
        int r = i / XPITCH, c = i % XPITCH;
        int rr = (r < 2) ? r : (r + 256);
        xs[rr * XPITCH + c] = 0;
    }
    __syncthreads();

    // ---- gather embeddings -> bf16 LDS (rows offset by +2 halo) ----
    // 256 tokens * 32 float4-chunks = 8192 chunks, 64 per thread
    #pragma unroll 4
    for (int i = 0; i < 64; ++i) {
        int c  = t + 128 * i;
        int l  = c >> 5;
        int e4 = c & 31;
        const float4 v = ((const float4*)embed)[(size_t)tok[l] * 32 + e4];
        uint2 pk;
        pk.x = (unsigned)f2bf(v.x) | ((unsigned)f2bf(v.y) << 16);
        pk.y = (unsigned)f2bf(v.z) | ((unsigned)f2bf(v.w) << 16);
        *(uint2*)(xs + (l + 2) * XPITCH + e4 * 4) = pk;
    }
    // ---- weights: conv_w[f][e][tap] -> wb[tap][f][e] bf16 ----
    #pragma unroll 4
    for (int i = 0; i < 160; ++i) {
        int idx = t + 128 * i;                // < 20480
        int e   = idx & 127;
        int f   = (idx >> 7) & 31;
        int tap = idx >> 12;
        wb[(tap * PF + f) * XPITCH + e] = f2bf(conv_w[(f * PE + e) * PW + tap]);
    }
    __syncthreads();

    // ---- conv as 5 shifted GEMMs: each wave owns 64 L-positions ----
    fvec8 zero = {0.f,0.f,0.f,0.f,0.f,0.f,0.f,0.f};
    fvec8 acc[4][2];
    #pragma unroll
    for (int mt = 0; mt < 4; ++mt) { acc[mt][0] = zero; acc[mt][1] = zero; }

    #pragma unroll
    for (int tap = 0; tap < PW; ++tap) {
        #pragma unroll
        for (int ks = 0; ks < 4; ++ks) {
            const int kb = ks * 32;
            // B fragments (K x N = 32x16), K contiguous per lane:
            // lane half selects K 0..15 / 16..31, N = mrow (+16 for tile 1)
            FragU b0, b1;
            {
                const unsigned short* p0 = wb + (tap * PF + mrow) * XPITCH + kb + half * 16;
                b0.q[0] = *(const uint4*)p0;
                b0.q[1] = *(const uint4*)(p0 + 8);
                const unsigned short* p1 = wb + (tap * PF + 16 + mrow) * XPITCH + kb + half * 16;
                b1.q[0] = *(const uint4*)p1;
                b1.q[1] = *(const uint4*)(p1 + 8);
            }
            #pragma unroll
            for (int mt = 0; mt < 4; ++mt) {
                // A fragment (M x K = 16x32): row = L + tap (halo absorbs -2)
                const int row = wave * 64 + mt * 16 + mrow + tap;
                const unsigned short* pa = xs + row * XPITCH + kb + half * 16;
                FragU a;
                a.q[0] = *(const uint4*)pa;          // K = kb + h16 .. +7
                a.q[1] = *(const uint4*)(pa + 8);    // K = kb + h16 + 8 .. +15
                acc[mt][0] = __builtin_amdgcn_wmma_f32_16x16x32_bf16(
                    false, a.v, false, b0.v, (short)0, acc[mt][0], false, false);
                acc[mt][1] = __builtin_amdgcn_wmma_f32_16x16x32_bf16(
                    false, a.v, false, b1.v, (short)0, acc[mt][1], false, false);
            }
        }
    }

    // ---- tanh + scatter C tiles to fm LDS (C layout: VGPR v -> M=v(+8)) ----
    #pragma unroll
    for (int mt = 0; mt < 4; ++mt)
        #pragma unroll
        for (int nt = 0; nt < 2; ++nt)
            #pragma unroll
            for (int v = 0; v < 8; ++v) {
                int r = wave * 64 + mt * 16 + v + half * 8;
                fm[r * FMPITCH + nt * 16 + mrow] = tanhf(acc[mt][nt][v]);
            }
    __syncthreads();

    // ---- reductions: thread (q, f) scans 64 L values of filter f ----
    {
        const int f = t & 31;
        const int q = t >> 5;
        float mx = -2.0f, s = 0.0f, tt = 0.0f;
        #pragma unroll 4
        for (int l = q * 64; l < q * 64 + 64; ++l) {
            float v = fm[l * FMPITCH + f];
            mx = fmaxf(mx, v);
            float e = __expf(v);     // v in [-1,1]: no max-shift needed
            s += e;
            tt += v * e;
        }
        pmax[t] = mx; ps[t] = s; pt[t] = tt;
    }
    __syncthreads();

    if (t < 32) {
        float mx = pmax[t], s = ps[t], tt = pt[t];
        #pragma unroll
        for (int q = 1; q < 4; ++q) {
            mx = fmaxf(mx, pmax[q * 32 + t]);
            s += ps[q * 32 + t];
            tt += pt[q * 32 + t];
        }
        maxbuf[(size_t)br * PF + t] = mx;
        float ent = __logf(s) - tt / s;      // row entropy
        #pragma unroll
        for (int o = 16; o > 0; o >>= 1) ent += __shfl_xor(ent, o, 32);
        if (t == 0) entbuf[br] = ent;
    }
}

// ---------------------------------------------------------------------------
// Kernel 2: blocks 0..B-1 -> ratings; block B -> entropy mean.
// ---------------------------------------------------------------------------
__global__ __launch_bounds__(64) void convpmf_finalize(
    const float* __restrict__ maxbuf,   // [PBR, PF]
    const float* __restrict__ entbuf,   // [PBR]
    const int*   __restrict__ uidx,     // [PB]
    const float* __restrict__ w_user,   // [NU, PF]
    const float* __restrict__ bias,     // [1]
    float* __restrict__ out)            // [PB + 1]
{
    const int b = blockIdx.x;
    const int t = threadIdx.x;
    if (b < PB) {
        if (t < 32) {
            float m = 0.0f;
            #pragma unroll
            for (int r = 0; r < PR; ++r)
                m += maxbuf[(size_t)(b * PR + r) * PF + t];
            m *= (1.0f / (float)PR);                       // mean over reviews
            float v = m * w_user[(size_t)uidx[b] * PF + t]; // user dot
            #pragma unroll
            for (int o = 16; o > 0; o >>= 1) v += __shfl_xor(v, o, 32);
            if (t == 0) out[b] = v + bias[0];
        }
    } else {
        __shared__ float red[64];
        float s = 0.0f;
        for (int i = t; i < PBR; i += 64) s += entbuf[i];
        red[t] = s;
        __syncthreads();
        #pragma unroll
        for (int w = 32; w > 0; w >>= 1) {
            if (t < w) red[t] += red[t + w];
            __syncthreads();
        }
        if (t == 0) out[PB] = red[0] / (float)(PBR * PF);
    }
}

// ---------------------------------------------------------------------------
extern "C" void kernel_launch(void* const* d_in, const int* in_sizes, int n_in,
                              void* d_out, int out_size, void* d_ws, size_t ws_size,
                              hipStream_t stream) {
    const int*   user_indices = (const int*)  d_in[0];   // [PB]
    const int*   docs         = (const int*)  d_in[1];   // [PB,PR,PL]
    const float* w_user       = (const float*)d_in[2];   // [NU,PF]
    const float* embed_table  = (const float*)d_in[3];   // [V,PE]
    const float* conv_w       = (const float*)d_in[4];   // [PF,PE,PW]
    const float* bias         = (const float*)d_in[5];   // [1]
    float* out = (float*)d_out;                           // [PB+1]

    float* maxbuf = (float*)d_ws;                         // PBR*PF floats
    float* entbuf = maxbuf + (size_t)PBR * PF;            // PBR floats

    // LDS: tok 1024 + xs 70720 + wb 43520 + fm 33792 + partials 1536
    const size_t shmem = 1024 + 70720 + 43520 + 33792 + 1536;   // 150592 B

    convpmf_main<<<dim3(PBR), dim3(128), shmem, stream>>>(
        docs, embed_table, conv_w, maxbuf, entbuf);

    convpmf_finalize<<<dim3(PB + 1), dim3(64), 0, stream>>>(
        maxbuf, entbuf, user_indices, w_user, bias, out);
}
